// MatchBatchGAT_18459769439025
// MI455X (gfx1250) — matrix operand order repaired and checked by reading, not compiled, not run on hardware
//
#include <hip/hip_runtime.h>
#include <hip/hip_bf16.h>
#include <math.h>

// Problem constants from the reference
#define BSZ   32
#define NN    512
#define NHEAD 8
#define FHID  16
#define FIN   64
#define FOUT  64

typedef __attribute__((ext_vector_type(16))) _Float16 v16h;
typedef __attribute__((ext_vector_type(8)))  float    v8f;
typedef __attribute__((ext_vector_type(4)))  float    v4f;
typedef __attribute__((ext_vector_type(4)))  int      v4i;
typedef __attribute__((ext_vector_type(4)))  unsigned v4u;

// ---------------------------------------------------------------------------
// K1: layer-1 typed projection hp1[b,h,n,o] = emb[b,n,:] @ w1[type,h,:,:]
//     plus a_s1/a_d1 = <tanh(hp1), a_src1/a_dst1>. hp1 stored f16 (WMMA B op).
// ---------------------------------------------------------------------------
__global__ __launch_bounds__(256) void gat1_proj(
    const float* __restrict__ features, const float* __restrict__ line_emb,
    const int* __restrict__ v_types, const float* __restrict__ w1,
    const float* __restrict__ a_src1, const float* __restrict__ a_dst1,
    _Float16* __restrict__ hp1, float* __restrict__ a_s1, float* __restrict__ a_d1)
{
    int gid = blockIdx.x * 256 + threadIdx.x;          // over BSZ*NHEAD*NN
    if (gid >= BSZ * NHEAD * NN) return;
    int n = gid & (NN - 1);
    int h = (gid >> 9) & (NHEAD - 1);
    int b = gid >> 12;

    int t = v_types[b * NN + n];
    const float* w    = w1 + ((long)(t * NHEAD + h) * FIN) * FHID;  // [f][o]
    const float* fptr = features + (long)(b * NN + n) * 32;
    const float* lptr = line_emb + (long)(b * NN + n) * 32;

    float acc[FHID];
#pragma unroll
    for (int o = 0; o < FHID; ++o) acc[o] = 0.f;

    for (int f = 0; f < FIN; ++f) {
        float x = (f < 32) ? fptr[f] : lptr[f - 32];
        const float* wr = w + f * FHID;
#pragma unroll
        for (int o = 0; o < FHID; ++o) acc[o] += x * wr[o];
    }

    float as = 0.f, ad = 0.f;
    long base = ((long)(b * NHEAD + h) * NN + n) * FHID;
#pragma unroll
    for (int o = 0; o < FHID; ++o) {
        hp1[base + o] = (_Float16)acc[o];
        float tt = tanhf(acc[o]);
        as += tt * a_src1[h * FHID + o];
        ad += tt * a_dst1[h * FHID + o];
    }
    a_s1[(b * NHEAD + h) * NN + n] = as;
    a_d1[(b * NHEAD + h) * NN + n] = ad;
}

// ---------------------------------------------------------------------------
// K2: fused layer-1 attention + softmax + attn1 output + WMMA aggregation.
// Block = 256 threads (8 waves) handles 16 rows for one (b,h).
// b128 loads/stores on the 512-wide sweeps; non-temporal attn1 stores so the
// 268MB output stream does not evict the L2-resident (33MB, 8x reused) adj.
// Wave 0 runs the 16x16x32 f16 WMMA K-loop over the tile in LDS.
// ---------------------------------------------------------------------------
__global__ __launch_bounds__(256) void gat1_attn_agg(
    const int* __restrict__ adj, const float* __restrict__ a_s1,
    const float* __restrict__ a_d1, const _Float16* __restrict__ hp1,
    const float* __restrict__ b1, float* __restrict__ attn_out,
    float* __restrict__ x1)
{
    __shared__ __align__(16) float    tile[16 * NN];  // logits -> exp -> attn
    __shared__ __align__(16) _Float16 hpl[NN * FHID]; // B operand (512x16 f16)
    __shared__ __align__(16) float    adL[NN];
    __shared__ float    asL[16];
    __shared__ float    red[16 * 16];
    __shared__ float    rowmax[16], rowinv[16];

    const int nt  = blockIdx.x;             // node tile (16 rows)
    const int h   = blockIdx.y;
    const int b   = blockIdx.z;
    const int tid = threadIdx.x;
    const int n0  = nt * 16;
    const int bh  = b * NHEAD + h;

    // Vectorized LDS fills (float4 / uint4 = b128)
    {
        const v4f* adSrc = (const v4f*)(a_d1 + (long)bh * NN);
        if (tid < NN / 4) ((v4f*)adL)[tid] = adSrc[tid];
        const v4u* hpSrc = (const v4u*)(hp1 + (long)bh * NN * FHID);
        v4u* hpDst = (v4u*)hpl;
#pragma unroll
        for (int s = 0; s < (NN * FHID / 8) / 256; ++s)
            hpDst[tid + s * 256] = hpSrc[tid + s * 256];
        if (tid < 16) asL[tid] = a_s1[bh * NN + n0 + tid];
    }
    __syncthreads();

    const int row = tid >> 4;                    // 16 threads per row
    const int j   = tid & 15;
    const int n   = n0 + row;
    const long adjbase = ((long)b * NN + n) * NN;
    const float as = asL[row];

    // Pass 1: masked leaky logits + row max (b128 adj loads, 4 elems/lane)
    float pmax = -INFINITY;
#pragma unroll
    for (int k = 0; k < 8; ++k) {
        int m0 = j * 4 + k * 64;
        v4i av  = *(const v4i*)(adj + adjbase + m0);
        v4f ad4 = *(const v4f*)(adL + m0);
        v4f lg;
#pragma unroll
        for (int q = 0; q < 4; ++q) {
            float e = as + ad4[q];
            e = (e >= 0.f) ? e : 0.2f * e;
            lg[q] = (av[q] != 0) ? e : -INFINITY;
            pmax = fmaxf(pmax, lg[q]);
        }
        *(v4f*)(tile + row * NN + m0) = lg;
    }
    red[row * 16 + j] = pmax;
    __syncthreads();
    if (tid < 16) {
        float mx = red[tid * 16];
        for (int q = 1; q < 16; ++q) mx = fmaxf(mx, red[tid * 16 + q]);
        rowmax[tid] = mx;
    }
    __syncthreads();

    // Pass 2: exp + row sum
    float mx = rowmax[row];
    float psum = 0.f;
#pragma unroll
    for (int k = 0; k < 8; ++k) {
        int m0 = j * 4 + k * 64;
        v4f lg = *(const v4f*)(tile + row * NN + m0);
        v4f p;
#pragma unroll
        for (int q = 0; q < 4; ++q) {
            p[q] = expf(lg[q] - mx);             // exp(-inf - mx) == 0
            psum += p[q];
        }
        *(v4f*)(tile + row * NN + m0) = p;
    }
    red[row * 16 + j] = psum;
    __syncthreads();
    if (tid < 16) {
        float s = 0.f;
        for (int q = 0; q < 16; ++q) s += red[tid * 16 + q];
        rowinv[tid] = 1.f / s;
    }
    __syncthreads();

    // Pass 3: normalize + coalesced NON-TEMPORAL b128 attn1 stores
    {
        float* outbase = attn_out + ((long)bh * NN + n0) * NN;
#pragma unroll
        for (int s = 0; s < 8; ++s) {
            int idx4 = (tid + s * 256) * 4;      // 0..8188, row-aligned
            int r = idx4 >> 9;
            float inv = rowinv[r];
            v4f a = *(const v4f*)(tile + idx4);
            a *= inv;
            *(v4f*)(tile + idx4) = a;
            __builtin_nontemporal_store(a, (v4f*)(outbase + idx4));
        }
    }
    __syncthreads();

    // Pass 4: wave 0 does D(16x16) = attn(16x512) @ hp1(512x16) via WMMA f16
    if (tid < 32) {
        int lane = tid;
        int g    = lane >> 4;
        int mrow = lane & 15;
        v8f c = {};
        for (int kk = 0; kk < 16; ++kk) {
            int Kb = kk * 32;
            v16h A, B;
            int kr = Kb + lane;                  // B row K for this lane
#pragma unroll
            for (int v = 0; v < 8; ++v) {
                // A 16x32 f16 layout (ISA 7.12.2): VGPR v, lane group g
                int kb = ((v & 4) ? 16 : 0) + g * 8 + (v & 3) * 2;
                A[2 * v]     = (_Float16)tile[mrow * NN + Kb + kb];
                A[2 * v + 1] = (_Float16)tile[mrow * NN + Kb + kb + 1];
                // B 32x16 f16: N = lane&15, K order same as A (kb, kb+1)
                B[2 * v]     = hpl[(Kb + kb) * FHID + mrow];
                B[2 * v + 1] = hpl[(Kb + kb + 1) * FHID + mrow];
            }
            c = __builtin_amdgcn_wmma_f32_16x16x32_f16(
                    false, A, false, B, (short)0, c, false, false);
        }
        // C/D f32 16x16 layout: lane<16 -> M=v, lane>=16 -> M=8+v; N = lane&15
        int o = lane & 15;
        float bb = b1[o];
#pragma unroll
        for (int v = 0; v < 8; ++v) {
            int M = (lane < 16) ? v : (8 + v);
            float val = c[v] + bb;
            val = (val > 0.f) ? val : (expf(val) - 1.f);   // elu
            // x1[b][n][h*16+o] (post transpose+reshape of reference)
            x1[((long)b * NN + n0 + M) * (NHEAD * FHID) + h * FHID + o] = val;
        }
    }
}

// ---------------------------------------------------------------------------
// K3: layer-2 typed projection hp2[b,n,:] = x1[b,n,:] @ w2[type,0,:,:]
//     plus a_s2/a_d2 tanh dots. Block = 64 threads = one node.
// ---------------------------------------------------------------------------
__global__ __launch_bounds__(64) void gat2_proj(
    const float* __restrict__ x1, const int* __restrict__ v_types,
    const float* __restrict__ w2, const float* __restrict__ a_src2,
    const float* __restrict__ a_dst2,
    float* __restrict__ hp2, float* __restrict__ a_s2, float* __restrict__ a_d2)
{
    int bn = blockIdx.x;                  // b*NN + n
    int o  = threadIdx.x;                 // 0..63
    int t  = v_types[bn];
    const float* xrow = x1 + (long)bn * (NHEAD * FHID);
    const float* w    = w2 + (long)t * (NHEAD * FHID) * FOUT + o;

    float acc = 0.f;
    for (int f = 0; f < NHEAD * FHID; ++f) acc += xrow[f] * w[(long)f * FOUT];
    hp2[(long)bn * FOUT + o] = acc;

    float tt = tanhf(acc);
    __shared__ float ps[64], pd[64];
    ps[o] = tt * a_src2[o];
    pd[o] = tt * a_dst2[o];
    __syncthreads();
    for (int s = 32; s > 0; s >>= 1) {
        if (o < s) { ps[o] += ps[o + s]; pd[o] += pd[o + s]; }
        __syncthreads();
    }
    if (o == 0) { a_s2[bn] = ps[0]; a_d2[bn] = pd[0]; }
}

// ---------------------------------------------------------------------------
// K4: layer-2 attention for rows 0,1 only + elu + sim-mul + MLP + log_softmax.
// One block per batch element.
// ---------------------------------------------------------------------------
__global__ __launch_bounds__(256) void gat2_head(
    const int* __restrict__ adj, const float* __restrict__ hp2,
    const float* __restrict__ a_s2, const float* __restrict__ a_d2,
    const float* __restrict__ b2,
    const float* __restrict__ fc1_w, const float* __restrict__ fc1_b,
    const float* __restrict__ fc2_w, const float* __restrict__ fc2_b,
    const float* __restrict__ fc3_w, const float* __restrict__ fc3_b,
    float* __restrict__ out_scores, float* __restrict__ out_vsim)
{
    int b   = blockIdx.x;
    int tid = threadIdx.x;
    __shared__ float adL[NN];
    __shared__ float attn2[2][NN];
    __shared__ float red[256];
    __shared__ float xr[2][FOUT];
    __shared__ float vsm[FOUT];
    __shared__ float v1[3 * FOUT];
    __shared__ float v2h[FOUT];
    __shared__ float sc[2];

    for (int i = tid; i < NN; i += 256) adL[i] = a_d2[b * NN + i];
    __syncthreads();

    for (int r = 0; r < 2; ++r) {
        float as = a_s2[b * NN + r];
        float pmax = -INFINITY;
        for (int k = 0; k < 2; ++k) {
            int m = tid + k * 256;
            int av = adj[((long)b * NN + r) * NN + m];
            float e = as + adL[m];
            e = (e >= 0.f) ? e : 0.2f * e;
            float lg = (av != 0) ? e : -INFINITY;
            attn2[r][m] = lg;
            pmax = fmaxf(pmax, lg);
        }
        red[tid] = pmax;
        __syncthreads();
        for (int s = 128; s > 0; s >>= 1) {
            if (tid < s) red[tid] = fmaxf(red[tid], red[tid + s]);
            __syncthreads();
        }
        float mx = red[0];
        __syncthreads();
        float psum = 0.f;
        for (int k = 0; k < 2; ++k) {
            int m = tid + k * 256;
            float p = expf(attn2[r][m] - mx);
            attn2[r][m] = p;
            psum += p;
        }
        red[tid] = psum;
        __syncthreads();
        for (int s = 128; s > 0; s >>= 1) {
            if (tid < s) red[tid] += red[tid + s];
            __syncthreads();
        }
        float inv = 1.f / red[0];
        __syncthreads();
        for (int k = 0; k < 2; ++k) attn2[r][tid + k * 256] *= inv;
        __syncthreads();
    }

    // out2 rows 0,1: (1x512) @ (512x64) + b2, then elu
    if (tid < 128) {
        int r = tid >> 6, o = tid & 63;
        float acc = b2[o];
        for (int m = 0; m < NN; ++m)
            acc += attn2[r][m] * hp2[((long)b * NN + m) * FOUT + o];
        xr[r][o] = (acc > 0.f) ? acc : (expf(acc) - 1.f);
    }
    __syncthreads();
    if (tid < FOUT) {
        float v = xr[0][tid] * xr[1][tid];
        vsm[tid] = v;
        out_vsim[b * FOUT + tid] = v;
    }
    __syncthreads();
    if (tid < 3 * FOUT) {
        float acc = fc1_b[tid];
        for (int k = 0; k < FOUT; ++k) acc += vsm[k] * fc1_w[k * (3 * FOUT) + tid];
        v1[tid] = fmaxf(acc, 0.f);
    }
    __syncthreads();
    if (tid < FOUT) {
        float acc = fc2_b[tid];
        for (int k = 0; k < 3 * FOUT; ++k) acc += v1[k] * fc2_w[k * FOUT + tid];
        v2h[tid] = fmaxf(acc, 0.f);
    }
    __syncthreads();
    if (tid < 2) {
        float acc = fc3_b[tid];
        for (int k = 0; k < FOUT; ++k) acc += v2h[k] * fc3_w[k * 2 + tid];
        sc[tid] = acc;
    }
    __syncthreads();
    if (tid == 0) {
        float m = fmaxf(sc[0], sc[1]);
        float lse = m + logf(expf(sc[0] - m) + expf(sc[1] - m));
        out_scores[b * 2 + 0] = sc[0] - lse;
        out_scores[b * 2 + 1] = sc[1] - lse;
    }
}

// ---------------------------------------------------------------------------
extern "C" void kernel_launch(void* const* d_in, const int* in_sizes, int n_in,
                              void* d_out, int out_size, void* d_ws, size_t ws_size,
                              hipStream_t stream) {
    const float* features = (const float*)d_in[0];
    const int*   adj      = (const int*)  d_in[1];
    // d_in[2]=bs, d_in[3]=num_v (compile-time), d_in[4]=svm_features (unused)
    const float* line_emb = (const float*)d_in[5];
    const int*   v_types  = (const int*)  d_in[6];
    const float* w1       = (const float*)d_in[7];
    const float* a_src1   = (const float*)d_in[8];
    const float* a_dst1   = (const float*)d_in[9];
    const float* b1       = (const float*)d_in[10];
    const float* w2       = (const float*)d_in[11];
    const float* a_src2   = (const float*)d_in[12];
    const float* a_dst2   = (const float*)d_in[13];
    const float* b2       = (const float*)d_in[14];
    const float* fc1_w    = (const float*)d_in[15];
    const float* fc1_b    = (const float*)d_in[16];
    const float* fc2_w    = (const float*)d_in[17];
    const float* fc2_b    = (const float*)d_in[18];
    const float* fc3_w    = (const float*)d_in[19];
    const float* fc3_b    = (const float*)d_in[20];

    float* out        = (float*)d_out;
    float* out_scores = out;                       // (32,2)
    float* out_vsim   = out + BSZ * 2;             // (32,64)
    float* out_attn1  = out + BSZ * 2 + BSZ * FOUT;// (32,8,512,512)

    char* ws = (char*)d_ws;
    _Float16* hp1  = (_Float16*)ws;                           // 4 MB
    float*    a_s1 = (float*)(ws + (4u  << 20));              // 512 KB
    float*    a_d1 = (float*)(ws + (4u  << 20) + (512u << 10));
    float*    x1   = (float*)(ws + (5u  << 20));              // 8 MB
    float*    hp2  = (float*)(ws + (13u << 20));              // 4 MB
    float*    a_s2 = (float*)(ws + (17u << 20));              // 64 KB
    float*    a_d2 = (float*)(ws + (17u << 20) + (64u << 10));

    gat1_proj<<<dim3((BSZ * NHEAD * NN) / 256), 256, 0, stream>>>(
        features, line_emb, v_types, w1, a_src1, a_dst1, hp1, a_s1, a_d1);

    gat1_attn_agg<<<dim3(NN / 16, NHEAD, BSZ), 256, 0, stream>>>(
        adj, a_s1, a_d1, hp1, b1, out_attn1, x1);

    gat2_proj<<<dim3(BSZ * NN), 64, 0, stream>>>(
        x1, v_types, w2, a_src2, a_dst2, hp2, a_s2, a_d2);

    gat2_head<<<dim3(BSZ), 256, 0, stream>>>(
        adj, hp2, a_s2, a_d2, b2, fc1_w, fc1_b, fc2_w, fc2_b, fc3_w, fc3_b,
        out_scores, out_vsim);
}
